// Attention_26886495273690
// MI455X (gfx1250) — matrix operand order repaired and checked by reading, not compiled, not run on hardware
//
#include <hip/hip_runtime.h>
#include <hip/hip_bf16.h>
#include <cstdint>
#include <cstddef>

typedef __attribute__((ext_vector_type(16))) __bf16 v16bf;
typedef __attribute__((ext_vector_type(8)))  __bf16 v8bf;
typedef __attribute__((ext_vector_type(4)))  __bf16 v4bf;
typedef __attribute__((ext_vector_type(8)))  float  v8f;
typedef __attribute__((ext_vector_type(4)))  float  v4f;
typedef __attribute__((ext_vector_type(4)))  unsigned int u32x4;
typedef __attribute__((ext_vector_type(8)))  int i32x8;
typedef __attribute__((ext_vector_type(4)))  int i32x4;

#define NHEAD   16
#define NLOCAL  4
#define HDIM    128
#define DMODEL  2048
#define BATCH   2
#define SEQ     2048
#define EQKV    3072   // (16 + 2*4) * 128
#define MTOT    (BATCH*SEQ)   // 4096

// LDS tile geometry for the TDM-staged GEMM:
// tile = 128 rows x 64 bf16 (128B) + 16B TDM pad per row -> 144B row stride
#define TROWS      128
#define TK         64
#define ROWSTRIDE  72            // elements (144 bytes)
#define TILE_E     (TROWS*ROWSTRIDE)      // 9216 elements = 18432 bytes
#define TILE_BYTES (TILE_E*2)
#define BUF_BYTES  (2*TILE_BYTES)         // A tile + B tile
#define SMEM_BYTES (2*BUF_BYTES)          // double buffered = 73728 B

// ---------------------------------------------------------------------------
// fp32 -> bf16 convert
// ---------------------------------------------------------------------------
__global__ __launch_bounds__(256) void k_tobf16(const float* __restrict__ in,
                                                __bf16* __restrict__ out, int n) {
  int i = blockIdx.x * 256 + threadIdx.x;
  if (i < n) out[i] = (__bf16)in[i];
}

// ---------------------------------------------------------------------------
// TDM: async-load a 128x64 bf16 tile (row stride K elements) into LDS at
// byte offset lds_off, padding 16B after each 128B row (D# pad feature).
// Descriptor layout per CDNA5 ISA section 8.3/8.4 (2-D tensor, groups 2/3 = 0).
// 6-arg builtin form (clang-23 / therock-10.0 toolchain).
// ---------------------------------------------------------------------------
__device__ __forceinline__ void tdm_load_tile(unsigned int lds_off,
                                              const __bf16* gptr,
                                              unsigned int rows_total,
                                              unsigned int K) {
  const unsigned long long ga = (unsigned long long)(uintptr_t)gptr;
  u32x4 g0;
  g0[0] = 1u;                                   // count=1, user mode
  g0[1] = lds_off;                              // lds_addr (bytes)
  g0[2] = (unsigned int)ga;                     // global_addr[31:0]
  g0[3] = ((unsigned int)(ga >> 32) & 0x01FFFFFFu) | (2u << 30); // addr[56:32] | type=2

  const unsigned int td0 = K;                   // tensor_dim0 (elements)
  const unsigned int td1 = rows_total;          // tensor_dim1
  i32x8 g1;
  // data_size=1 (2B) | pad_enable | pad_interval=4 (32 dwords) | pad_amount=3 (4 dwords)
  g1[0] = (int)((1u << 16) | (1u << 20) | (4u << 22) | (3u << 25));
  g1[1] = (int)((td0 & 0xFFFFu) << 16);                 // tensor_dim0[15:0]
  g1[2] = (int)((td0 >> 16) | ((td1 & 0xFFFFu) << 16)); // td0 hi | td1 lo
  g1[3] = (int)((td1 >> 16) | ((unsigned)TK << 16));    // td1 hi | tile_dim0=64
  g1[4] = (int)TROWS;                                   // tile_dim1=128, tile_dim2=0
  g1[5] = (int)K;                                       // tensor_dim0_stride[31:0]
  g1[6] = 0;                                            // stride hi, dim1_stride lo
  g1[7] = 0;
  i32x4 g2 = {0, 0, 0, 0};
  i32x4 g3 = {0, 0, 0, 0};
  i32x8 g4 = {0, 0, 0, 0, 0, 0, 0, 0};
  __builtin_amdgcn_tensor_load_to_lds(g0, g1, g2, g3, g4, 0);
}

__device__ __forceinline__ v16bf pack16(const __bf16* p, int hioff) {
  v8bf lo = *(const v8bf*)p;
  v8bf hi = *(const v8bf*)(p + hioff);
  v16bf r;
#pragma unroll
  for (int i = 0; i < 8; ++i) { r[i] = lo[i]; r[8 + i] = hi[i]; }
  return r;
}

// ---------------------------------------------------------------------------
// C[M,N] = A[M,K] * B[N,K]^T with TDM-staged LDS tiles.
// Block = 256 threads (8 waves) -> 128M x 128N tile; wave (wm 0..3, wn 0..1)
// owns 32M x 64N (2x4 accumulators). K-step 64, double buffered.
// ---------------------------------------------------------------------------
__global__ __launch_bounds__(256) void k_gemm_bf16(const __bf16* __restrict__ A,
                                                   const __bf16* __restrict__ Bw,
                                                   float* __restrict__ C,
                                                   int M, int N, int K) {
  extern __shared__ __bf16 smem[];   // only dynamic LDS in this kernel -> offset 0
  const int lane = threadIdx.x & 31;
  const int wave = threadIdx.x >> 5;
  const int l15  = lane & 15;
  const int hi16 = lane >> 4;
  const int wm   = wave & 3;         // M quadrant
  const int wn   = wave >> 2;        // N half
  const int m_off = wm * 32;
  const int n_off = wn * 64;
  const int mblk = blockIdx.y * 128;
  const int nblk = blockIdx.x * 128;
  const int aklo = hi16 ? 8 : 0;
  const int bklo = hi16 ? 8 : 0;

  v8f acc[2][4] = {};

  const int nk = K / TK;
  if (wave == 0) {
    tdm_load_tile(0u,            A  + (size_t)mblk * K, (unsigned)M, (unsigned)K);
    tdm_load_tile(TILE_BYTES,    Bw + (size_t)nblk * K, (unsigned)N, (unsigned)K);
  }

  for (int kt = 0; kt < nk; ++kt) {
    const int cur = kt & 1;
    if (wave == 0) {
      if (kt + 1 < nk) {
        const unsigned int nb = (unsigned int)((cur ^ 1) * BUF_BYTES);
        const int knext = (kt + 1) * TK;
        tdm_load_tile(nb,              A  + (size_t)mblk * K + knext, (unsigned)M, (unsigned)K);
        tdm_load_tile(nb + TILE_BYTES, Bw + (size_t)nblk * K + knext, (unsigned)N, (unsigned)K);
        __builtin_amdgcn_s_wait_tensorcnt((short)2);   // current buffer's 2 done
      } else {
        __builtin_amdgcn_s_wait_tensorcnt((short)0);
      }
    }
    __syncthreads();

    const __bf16* la = smem + (size_t)cur * (BUF_BYTES / 2);
    const __bf16* lb = la + TILE_E;
#pragma unroll
    for (int kk = 0; kk < TK; kk += 32) {
      v16bf af[2];
#pragma unroll
      for (int t = 0; t < 2; ++t)
        af[t] = pack16(la + (size_t)(m_off + t * 16 + l15) * ROWSTRIDE + kk + aklo, 16);
#pragma unroll
      for (int u = 0; u < 4; ++u) {
        v16bf bf = pack16(lb + (size_t)(n_off + u * 16 + l15) * ROWSTRIDE + kk + bklo, 16);
#pragma unroll
        for (int t = 0; t < 2; ++t)
          acc[t][u] = __builtin_amdgcn_wmma_f32_16x16x32_bf16(
              false, af[t], false, bf, (short)0, acc[t][u], false, false);
      }
    }
    __syncthreads();   // protect buffer before next TDM overwrite
  }

#pragma unroll
  for (int t = 0; t < 2; ++t) {
    const int rbase = mblk + m_off + t * 16 + (hi16 ? 8 : 0);
#pragma unroll
    for (int u = 0; u < 4; ++u)
#pragma unroll
      for (int r = 0; r < 8; ++r)
        C[(size_t)(rbase + r) * N + nblk + n_off + u * 16 + l15] = acc[t][u][r];
  }
}

// ---------------------------------------------------------------------------
// RMSNorm + RoPE. One wave per (b, s, head). fp32 in, bf16 out[b][h][s][128].
// ---------------------------------------------------------------------------
__global__ __launch_bounds__(128) void k_norm_rope(const float* __restrict__ qkv,
                                                   const float* __restrict__ w,
                                                   const float* __restrict__ freqs,
                                                   __bf16* __restrict__ out,
                                                   int colbase, int H) {
  const int lane = threadIdx.x & 31;
  const int wv   = threadIdx.x >> 5;
  const int id   = blockIdx.x * 4 + wv;
  const int h = id % H;
  const int s = (id / H) % SEQ;
  const int b = id / (H * SEQ);
  if (b >= BATCH) return;

  const float* src = qkv + ((size_t)(b * SEQ + s)) * EQKV + colbase + h * HDIM + lane * 4;
  v4f x = *(const v4f*)src;
  float ss = x[0]*x[0] + x[1]*x[1] + x[2]*x[2] + x[3]*x[3];
#pragma unroll
  for (int off = 1; off < 32; off <<= 1) ss += __shfl_xor(ss, off, 32);
  const float rn = rsqrtf(ss * (1.0f / (float)HDIM) + 1e-5f);

  v4f wv4 = *(const v4f*)(w + lane * 4);
  float xn[4];
#pragma unroll
  for (int j = 0; j < 4; ++j) xn[j] = x[j] * rn * wv4[j];

  v4bf ov;
#pragma unroll
  for (int p = 0; p < 2; ++p) {
    const int fi = s * (HDIM / 2) + lane * 2 + p;
    const float cr = freqs[fi * 2 + 0];
    const float ci = freqs[fi * 2 + 1];
    const float xr = xn[2 * p], xi = xn[2 * p + 1];
    ov[2 * p]     = (__bf16)(xr * cr - xi * ci);
    ov[2 * p + 1] = (__bf16)(xi * cr + xr * ci);
  }
  __bf16* dst = out + (((size_t)(b * H + h)) * SEQ + s) * HDIM + lane * 4;
  *(v4bf*)dst = ov;
}

// ---------------------------------------------------------------------------
// V transpose: vt[b][hkv][d][s] = bf16(qkv[b][s][2560 + hkv*128 + d])
// ---------------------------------------------------------------------------
__global__ __launch_bounds__(256) void k_vtrans(const float* __restrict__ qkv,
                                                __bf16* __restrict__ vt) {
  const int i = blockIdx.x * 256 + threadIdx.x;
  const int total = BATCH * NLOCAL * HDIM * SEQ;
  if (i >= total) return;
  const int s  = i % SEQ;
  const int d  = (i / SEQ) % HDIM;
  const int hh = (i / (SEQ * HDIM)) % NLOCAL;
  const int b  = i / (SEQ * HDIM * NLOCAL);
  vt[i] = (__bf16)qkv[((size_t)(b * SEQ + s)) * EQKV + NHEAD * HDIM + NLOCAL * HDIM + hh * HDIM + d];
}

// ---------------------------------------------------------------------------
// Flash-style causal attention. grid = (SEQ/64, BATCH*NHEAD), block = 128.
// ---------------------------------------------------------------------------
__global__ __launch_bounds__(128) void k_attn(const __bf16* __restrict__ qb,
                                              const __bf16* __restrict__ kb,
                                              const __bf16* __restrict__ vt,
                                              __bf16* __restrict__ yb) {
  __shared__ __bf16 lds_p[4][16 * 32];
  const int lane = threadIdx.x & 31;
  const int wave = threadIdx.x >> 5;
  const int l15  = lane & 15;
  const int hi16 = lane >> 4;
  const int bh = blockIdx.y;
  const int b = bh >> 4, h = bh & 15, hkv = h >> 2;
  const int q0 = blockIdx.x * 64 + wave * 16;
  const int aklo = hi16 ? 8 : 0;
  const float scale = 0.08838834764831845f;   // 1/sqrt(128)

  const __bf16* qrow = qb + (((size_t)(b * NHEAD + h)) * SEQ + q0 + l15) * HDIM;
  v16bf qf[4];
#pragma unroll
  for (int c = 0; c < 4; ++c)
    qf[c] = pack16(qrow + c * 32 + aklo, 16);

  v8f o[8] = {};
  float mrow[8], lrow[8];
#pragma unroll
  for (int r = 0; r < 8; ++r) { mrow[r] = -3.0e38f; lrow[r] = 0.0f; }

  const __bf16* kbase = kb + ((size_t)(b * NLOCAL + hkv)) * SEQ * HDIM;
  const __bf16* vbase = vt + ((size_t)(b * NLOCAL + hkv)) * HDIM * SEQ;

  for (int k0 = 0; k0 < q0 + 16; k0 += 32) {
    v8f st[2] = {};
#pragma unroll
    for (int t = 0; t < 2; ++t) {
      const __bf16* krow = kbase + (size_t)(k0 + t * 16 + l15) * HDIM;
#pragma unroll
      for (int c = 0; c < 4; ++c) {
        v16bf bf = pack16(krow + c * 32 + hi16 * 8, 16);
        st[t] = __builtin_amdgcn_wmma_f32_16x16x32_bf16(
            false, qf[c], false, bf, (short)0, st[t], false, false);
      }
    }
#pragma unroll
    for (int t = 0; t < 2; ++t) {
      const int key = k0 + t * 16 + l15;
#pragma unroll
      for (int r = 0; r < 8; ++r) {
        const int qi = q0 + r + hi16 * 8;
        const float sv = st[t][r] * scale;
        st[t][r] = (key <= qi) ? sv : -1.0e30f;
      }
    }
    float lm[8];
#pragma unroll
    for (int r = 0; r < 8; ++r) lm[r] = fmaxf(st[0][r], st[1][r]);
#pragma unroll
    for (int off = 1; off < 16; off <<= 1)
#pragma unroll
      for (int r = 0; r < 8; ++r) lm[r] = fmaxf(lm[r], __shfl_xor(lm[r], off, 32));
    float alpha[8];
#pragma unroll
    for (int r = 0; r < 8; ++r) {
      const float mn = fmaxf(mrow[r], lm[r]);
      alpha[r] = __expf(mrow[r] - mn);
      mrow[r] = mn;
    }
    float ps[8];
#pragma unroll
    for (int r = 0; r < 8; ++r) {
      st[0][r] = __expf(st[0][r] - mrow[r]);
      st[1][r] = __expf(st[1][r] - mrow[r]);
      ps[r] = st[0][r] + st[1][r];
    }
#pragma unroll
    for (int off = 1; off < 16; off <<= 1)
#pragma unroll
      for (int r = 0; r < 8; ++r) ps[r] += __shfl_xor(ps[r], off, 32);
#pragma unroll
    for (int r = 0; r < 8; ++r) lrow[r] = lrow[r] * alpha[r] + ps[r];
#pragma unroll
    for (int d = 0; d < 8; ++d)
#pragma unroll
      for (int r = 0; r < 8; ++r) o[d][r] *= alpha[r];

    __bf16* pl = &lds_p[wave][0];
#pragma unroll
    for (int t = 0; t < 2; ++t)
#pragma unroll
      for (int r = 0; r < 8; ++r)
        pl[(r + hi16 * 8) * 32 + t * 16 + l15] = (__bf16)st[t][r];
    asm volatile("s_wait_dscnt 0x0" ::: "memory");

    v16bf pf = pack16(pl + l15 * 32 + aklo, 16);

#pragma unroll
    for (int d = 0; d < 8; ++d) {
      const __bf16* vrow = vbase + (size_t)(d * 16 + l15) * SEQ + k0 + hi16 * 8;
      v16bf vf = pack16(vrow, 16);
      o[d] = __builtin_amdgcn_wmma_f32_16x16x32_bf16(
          false, pf, false, vf, (short)0, o[d], false, false);
    }
  }

  float rinv[8];
#pragma unroll
  for (int r = 0; r < 8; ++r) rinv[r] = 1.0f / lrow[r];
#pragma unroll
  for (int d = 0; d < 8; ++d)
#pragma unroll
    for (int r = 0; r < 8; ++r) {
      const int row = q0 + r + hi16 * 8;
      yb[((size_t)b * SEQ + row) * DMODEL + h * HDIM + d * 16 + l15] =
          (__bf16)(o[d][r] * rinv[r]);
    }
}

// ---------------------------------------------------------------------------
// host-side launch
// ---------------------------------------------------------------------------
extern "C" void kernel_launch(void* const* d_in, const int* in_sizes, int n_in,
                              void* d_out, int out_size, void* d_ws, size_t ws_size,
                              hipStream_t stream) {
  const float* x      = (const float*)d_in[0];
  const float* freqs  = (const float*)d_in[1];
  // d_in[2] = mask (unused; causal mask applied analytically)
  const float* wqkv   = (const float*)d_in[3];
  const float* wo     = (const float*)d_in[4];
  const float* qnw    = (const float*)d_in[5];
  const float* knw    = (const float*)d_in[6];
  float* out = (float*)d_out;

  char* ws = (char*)d_ws;
  size_t off = 0;
  auto alloc = [&](size_t bytes) { size_t p = off; off = (off + bytes + 255) & ~(size_t)255; return p; };
  __bf16* xb    = (__bf16*)(ws + alloc((size_t)MTOT * DMODEL * 2));
  __bf16* wqkvb = (__bf16*)(ws + alloc((size_t)EQKV * DMODEL * 2));
  __bf16* wob   = (__bf16*)(ws + alloc((size_t)DMODEL * DMODEL * 2));
  float*  qkvf  = (float*) (ws + alloc((size_t)MTOT * EQKV * 4));
  __bf16* qb    = (__bf16*)(ws + alloc((size_t)BATCH * NHEAD  * SEQ * HDIM * 2));
  __bf16* kb    = (__bf16*)(ws + alloc((size_t)BATCH * NLOCAL * SEQ * HDIM * 2));
  __bf16* vt    = (__bf16*)(ws + alloc((size_t)BATCH * NLOCAL * HDIM * SEQ * 2));
  __bf16* yb    = (__bf16*)(ws + alloc((size_t)MTOT * DMODEL * 2));

  {
    int n = MTOT * DMODEL;
    k_tobf16<<<(n + 255) / 256, 256, 0, stream>>>(x, xb, n);
    n = EQKV * DMODEL;
    k_tobf16<<<(n + 255) / 256, 256, 0, stream>>>(wqkv, wqkvb, n);
    n = DMODEL * DMODEL;
    k_tobf16<<<(n + 255) / 256, 256, 0, stream>>>(wo, wob, n);
  }
  // qkv = x @ wqkv^T   (M=4096, N=3072, K=2048), TDM-staged tiles
  k_gemm_bf16<<<dim3(EQKV / 128, MTOT / 128), 256, SMEM_BYTES, stream>>>(
      xb, wqkvb, qkvf, MTOT, EQKV, DMODEL);
  k_norm_rope<<<(BATCH * SEQ * NHEAD) / 4, 128, 0, stream>>>(
      qkvf, qnw, freqs, qb, 0, NHEAD);
  k_norm_rope<<<(BATCH * SEQ * NLOCAL) / 4, 128, 0, stream>>>(
      qkvf, knw, freqs, kb, NHEAD * HDIM, NLOCAL);
  {
    int n = BATCH * NLOCAL * HDIM * SEQ;
    k_vtrans<<<(n + 255) / 256, 256, 0, stream>>>(qkvf, vt);
  }
  k_attn<<<dim3(SEQ / 64, BATCH * NHEAD), 128, 0, stream>>>(qb, kb, vt, yb);
  // out = y @ wo^T  (M=4096, N=2048, K=2048) -> fp32 d_out
  k_gemm_bf16<<<dim3(DMODEL / 128, MTOT / 128), 256, SMEM_BYTES, stream>>>(
      yb, wob, out, MTOT, DMODEL, DMODEL);
  (void)in_sizes; (void)n_in; (void)out_size; (void)ws_size;
}
